// QuantizedSparseLinear_15908558864459
// MI455X (gfx1250) — compile-verified, matrix operand order checked by reading, not run on hardware
//
#include <hip/hip_runtime.h>

// Sizes fixed by the reference problem.
#define IN_F   4096
#define OUT_F  4096
#define NTOK   8192
#define NBLK   2621
#define BLK    256

#define KSTEP  32          // K per stage (one WMMA K-depth)
#define TILE_M 128
#define TILE_N 256
#define NSTAGE (IN_F / KSTEP)

typedef __attribute__((ext_vector_type(16))) _Float16     v16h;
typedef __attribute__((ext_vector_type(8)))  _Float16     v8h;
typedef __attribute__((ext_vector_type(8)))  float        v8f;
typedef __attribute__((ext_vector_type(4)))  float        v4f;
typedef __attribute__((ext_vector_type(4)))  unsigned int v4u;
typedef __attribute__((ext_vector_type(8)))  int          v8i;
typedef __attribute__((ext_vector_type(4)))  int          v4i;

// ---------------------------------------------------------------------------
// Kernel 1: zero the dense f16 weight buffer (32 MB) with 16B stores.
// ---------------------------------------------------------------------------
__global__ __launch_bounds__(256) void qsl_zero_w(v4u* __restrict__ p, long n16) {
  long i = (long)blockIdx.x * blockDim.x + threadIdx.x;
  if (i < n16) {
    v4u z = {0u, 0u, 0u, 0u};
    p[i] = z;
  }
}

// ---------------------------------------------------------------------------
// Kernel 2: dequantize 4-bit codes and scatter into the dense f16 W.
// unpack_4bit: even position = high nibble, odd = low nibble.
// ---------------------------------------------------------------------------
__global__ __launch_bounds__(256) void qsl_scatter(
    const unsigned char* __restrict__ packed,   // [NBLK][BLK/2]
    const float* __restrict__ scales,           // [NBLK]
    const float* __restrict__ zps,              // [NBLK]
    const int* __restrict__ indices,            // [NBLK][BLK]
    _Float16* __restrict__ W)                   // [OUT_F*IN_F]
{
  const int b = blockIdx.x;
  const int p = threadIdx.x;
  const unsigned char byte = packed[b * (BLK / 2) + (p >> 1)];
  const int q = (p & 1) ? (byte & 0xF) : (byte >> 4);
  const float v = (float)q * scales[b] + zps[b];
  const int idx = indices[b * BLK + p];
  W[idx] = (_Float16)v;
}

// ---------------------------------------------------------------------------
// Kernel 3: convert activations f32 -> f16 (8 elements per thread).
// ---------------------------------------------------------------------------
__global__ __launch_bounds__(256) void qsl_cvt_x(
    const float* __restrict__ x, _Float16* __restrict__ xh, long n8)
{
  long i = (long)blockIdx.x * blockDim.x + threadIdx.x;
  if (i >= n8) return;
  const v4f* px = (const v4f*)(x + i * 8);
  v4f a = px[0];
  v4f b = px[1];
  v8h h;
  h[0] = (_Float16)a[0]; h[1] = (_Float16)a[1];
  h[2] = (_Float16)a[2]; h[3] = (_Float16)a[3];
  h[4] = (_Float16)b[0]; h[5] = (_Float16)b[1];
  h[6] = (_Float16)b[2]; h[7] = (_Float16)b[3];
  *(v8h*)(xh + i * 8) = h;
}

// ---------------------------------------------------------------------------
// TDM helper: issue a 2D tile load (tileRows x KSTEP f16 elements) from a
// row-major [tensorRows][IN_F] f16 tensor into LDS at byte address ldsAddr.
// Descriptor layout per CDNA5 ISA 8.3-8.6 (D# groups 0-3):
//   g0: count=1 | lds_addr | global_addr[56:0] | type=2
//   g1: data_size=1(2B); tensor_dim0=IN_F; tensor_dim1=tensorRows;
//       tile_dim0=KSTEP; tile_dim1=tileRows; tensor_dim0_stride=IN_F
//   g2/g3: zero (2D tile; tile_dim2/3/4 = 0 -> unused)
// ---------------------------------------------------------------------------
__device__ __forceinline__ void qsl_tdm_load_tile(
    const _Float16* gtile, unsigned ldsAddr,
    unsigned tensorRows, unsigned tileRows)
{
  unsigned long long ga = (unsigned long long)(uintptr_t)gtile;
  v4u g0;
  g0[0] = 1u;                                            // count=1
  g0[1] = ldsAddr;                                       // lds_addr (bytes)
  g0[2] = (unsigned)ga;                                  // global_addr[31:0]
  g0[3] = (unsigned)((ga >> 32) & 0x01FFFFFFull)         // global_addr[56:32]
          | (2u << 30);                                  // type=2 ("image")
  v8i g1;
  g1[0] = (int)(1u << 16);                               // data_size = 2 bytes
  g1[1] = (int)(((unsigned)IN_F & 0xFFFFu) << 16);       // tensor_dim0[15:0]
  g1[2] = (int)((((unsigned)IN_F >> 16) & 0xFFFFu)       // tensor_dim0[31:16]
          | ((tensorRows & 0xFFFFu) << 16));             // tensor_dim1[15:0]
  g1[3] = (int)(((tensorRows >> 16) & 0xFFFFu)           // tensor_dim1[31:16]
          | ((unsigned)KSTEP << 16));                    // tile_dim0
  g1[4] = (int)(tileRows & 0xFFFFu);                     // tile_dim1 (tile_dim2=0)
  g1[5] = (int)IN_F;                                     // tensor_dim0_stride[31:0]
  g1[6] = 0;                                             // stride[47:32], d1stride lo
  g1[7] = 0;
  v4i g2 = {0, 0, 0, 0};
  v4i g3 = {0, 0, 0, 0};
#if defined(__clang_major__) && (__clang_major__ >= 23)
  __builtin_amdgcn_tensor_load_to_lds(g0, g1, g2, g3, g1, 0);   // 6-arg toolchain
#else
  __builtin_amdgcn_tensor_load_to_lds(g0, g1, g2, g3, 0);       // ROCm 7.2 (clang-22)
#endif
}

// ---------------------------------------------------------------------------
// Kernel 4: dense GEMM  Y[t,o] = sum_k Xh[t,k] * Wh[o,k] + bias[o]
// TDM double-buffered LDS staging + v_wmma_f32_16x16x32_f16.
//
// Block = 256 threads (8 wave32), grid = (OUT_F/TILE_N, NTOK/TILE_M).
// Waves arranged 2(M) x 4(N); each wave computes 64x64 as 4x4 WMMA tiles.
// Per stage: wave0 DMAs A tile (128x32), wave1 DMAs B tile (256x32) into the
// spare LDS buffer; issuing waves s_wait_tensorcnt, then workgroup barrier.
// Fragments: two aligned ds_load_b128 per operand per the 16-bit register
// layout (lane row = lane&15; K runs at koff and koff+16, koff = 8*(lane>=16)).
// ---------------------------------------------------------------------------
__global__ __launch_bounds__(256) void qsl_gemm_wmma(
    const _Float16* __restrict__ X,    // [NTOK][IN_F] f16
    const _Float16* __restrict__ W,    // [OUT_F][IN_F] f16 dense
    const float* __restrict__ bias,    // [OUT_F]
    float* __restrict__ Y)             // [NTOK][OUT_F] f32
{
  __shared__ _Float16 ldsA[2][TILE_M * KSTEP];   // 2 x  8 KB
  __shared__ _Float16 ldsB[2][TILE_N * KSTEP];   // 2 x 16 KB

  const int lane  = threadIdx.x & 31;
  const int wave  = threadIdx.x >> 5;
  const int waveM = wave >> 2;                   // 0..1
  const int waveN = wave & 3;                    // 0..3
  const int mBase = blockIdx.y * TILE_M;         // block tile origin (M)
  const int nBase = blockIdx.x * TILE_N;         // block tile origin (N)
  const int lrow  = lane & 15;
  const int koff  = (lane >> 4) << 3;            // 0 or 8 halves

  v8f zero = {};
  v8f acc[4][4];
#pragma unroll
  for (int mi = 0; mi < 4; ++mi)
#pragma unroll
    for (int ni = 0; ni < 4; ++ni) acc[mi][ni] = zero;

  // Prologue: stage 0 into buffer 0.
  if (wave == 0) {
    qsl_tdm_load_tile(X + (long)mBase * IN_F, (unsigned)(uintptr_t)&ldsA[0][0],
                      NTOK, TILE_M);
  } else if (wave == 1) {
    qsl_tdm_load_tile(W + (long)nBase * IN_F, (unsigned)(uintptr_t)&ldsB[0][0],
                      OUT_F, TILE_N);
  }

  const int aRow0 = waveM * 64 + lrow;           // within A tile
  const int bRow0 = waveN * 64 + lrow;           // within B tile

  for (int s = 0; s < NSTAGE; ++s) {
    const int buf = s & 1;

    if (s + 1 < NSTAGE) {
      // Overlap: DMA next stage into the spare buffer while computing.
      const int k1 = (s + 1) * KSTEP;
      if (wave == 0) {
        qsl_tdm_load_tile(X + (long)mBase * IN_F + k1,
                          (unsigned)(uintptr_t)&ldsA[buf ^ 1][0], NTOK, TILE_M);
      } else if (wave == 1) {
        qsl_tdm_load_tile(W + (long)nBase * IN_F + k1,
                          (unsigned)(uintptr_t)&ldsB[buf ^ 1][0], OUT_F, TILE_N);
      }
      __builtin_amdgcn_s_wait_tensorcnt(1);      // stage s complete (in-order)
    } else {
      __builtin_amdgcn_s_wait_tensorcnt(0);      // final stage complete
    }
    __syncthreads();                             // tile visible to all waves

    v16h a[4], b[4];
#pragma unroll
    for (int mi = 0; mi < 4; ++mi) {
      const _Float16* p = &ldsA[buf][(aRow0 + mi * 16) * KSTEP + koff];
      v8h lo = *(const v8h*)p;
      v8h hi = *(const v8h*)(p + 16);
      a[mi] = __builtin_shufflevector(lo, hi,
               0, 1, 2, 3, 4, 5, 6, 7, 8, 9, 10, 11, 12, 13, 14, 15);
    }
#pragma unroll
    for (int ni = 0; ni < 4; ++ni) {
      const _Float16* p = &ldsB[buf][(bRow0 + ni * 16) * KSTEP + koff];
      v8h lo = *(const v8h*)p;
      v8h hi = *(const v8h*)(p + 16);
      b[ni] = __builtin_shufflevector(lo, hi,
               0, 1, 2, 3, 4, 5, 6, 7, 8, 9, 10, 11, 12, 13, 14, 15);
    }

#pragma unroll
    for (int mi = 0; mi < 4; ++mi)
#pragma unroll
      for (int ni = 0; ni < 4; ++ni)
        acc[mi][ni] = __builtin_amdgcn_wmma_f32_16x16x32_f16(
            false, a[mi], false, b[ni], (short)0, acc[mi][ni], false, false);

    __syncthreads();   // all reads of buf done before it is DMA-overwritten
  }

  // C/D layout: VGPR r -> M = r + 8*(lane>=16); N = lane & 15.
  const int mW   = mBase + waveM * 64;
  const int nW   = nBase + waveN * 64;
  const int mOff = (lane >> 4) << 3;
  const int ncol = lane & 15;
#pragma unroll
  for (int ni = 0; ni < 4; ++ni) {
    const float bv = bias[nW + ni * 16 + ncol];
#pragma unroll
    for (int mi = 0; mi < 4; ++mi) {
#pragma unroll
      for (int r = 0; r < 8; ++r) {
        Y[(long)(mW + mi * 16 + mOff + r) * OUT_F + (nW + ni * 16 + ncol)]
            = acc[mi][ni][r] + bv;
      }
    }
  }
}

// ---------------------------------------------------------------------------
// Launcher. Workspace layout:
//   [0, 32MB)   : dense W f16  (OUT_F*IN_F*2 bytes)
//   [32MB,96MB) : X f16        (NTOK*IN_F*2 bytes)
// ---------------------------------------------------------------------------
extern "C" void kernel_launch(void* const* d_in, const int* in_sizes, int n_in,
                              void* d_out, int out_size, void* d_ws, size_t ws_size,
                              hipStream_t stream) {
  const float*         x       = (const float*)d_in[0];
  const unsigned char* packed  = (const unsigned char*)d_in[1];
  const float*         scales  = (const float*)d_in[2];
  const float*         zps     = (const float*)d_in[3];
  const int*           indices = (const int*)d_in[4];
  const float*         bias    = (const float*)d_in[5];
  float*               out     = (float*)d_out;

  _Float16* Wf16 = (_Float16*)d_ws;
  _Float16* Xf16 = (_Float16*)((char*)d_ws + (size_t)OUT_F * IN_F * sizeof(_Float16));

  {
    long n16 = (long)OUT_F * IN_F * sizeof(_Float16) / 16;   // 2,097,152
    qsl_zero_w<<<(int)(n16 / 256), 256, 0, stream>>>((v4u*)Wf16, n16);
  }
  qsl_scatter<<<NBLK, BLK, 0, stream>>>(packed, scales, zps, indices, Wf16);
  {
    long n8 = (long)NTOK * IN_F / 8;                          // 4,194,304
    qsl_cvt_x<<<(int)(n8 / 256), 256, 0, stream>>>(x, Xf16, n8);
  }
  {
    dim3 grid(OUT_F / TILE_N, NTOK / TILE_M);                 // (16, 64)
    qsl_gemm_wmma<<<grid, 256, 0, stream>>>(Xf16, Wf16, bias, out);
  }
}
